// EdgeConvTongzhou_2508260901517
// MI455X (gfx1250) — matrix operand order repaired and checked by reading, not compiled, not run on hardware
//
#include <hip/hip_runtime.h>
#include <hip/hip_bf16.h>

typedef _Float16 v8h  __attribute__((ext_vector_type(8)));
typedef _Float16 v16h __attribute__((ext_vector_type(16)));
typedef float    v8f  __attribute__((ext_vector_type(8)));

#define NT   8      // 8 x 16 = 128 output channels (M tiles, one GN group each)
#define EPB  128    // edges/nodes per block (8 waves x 16 rows)

// ---------------- d_in index map (jax pytree alphabetical flatten) ----------
enum {
  IN_X = 0, IN_POS, IN_EI, IN_BATCH, IN_BS,
  ENC_B1 = 5, ENC_B2, ENC_B3, ENC_BE1, ENC_BE2, ENC_G1, ENC_G2, ENC_W1, ENC_W2, ENC_W3,
  E2_BE1 = 15, E2_BE2, E2_G1, E2_G2, E2_W1, E2_W2,
  ENCP_B1 = 21, ENCP_B2, ENCP_B3, ENCP_BE1, ENCP_BE2, ENCP_G1, ENCP_G2, ENCP_W1, ENCP_W2, ENCP_W3,
  FC_B1 = 31, FC_B2, FC_B3, FC_W1, FC_W2, FC_W3,
  LOC_BE1 = 37, LOC_BE2, LOC_BE3, LOC_G1, LOC_G2, LOC_G3, LOC_W1, LOC_W2, LOC_W3
};

// ---------------- helpers ---------------------------------------------------
__device__ __forceinline__ v8f zero8() {
  v8f z;
  #pragma unroll
  for (int i = 0; i < 8; ++i) z[i] = 0.0f;
  return z;
}

__device__ __forceinline__ v16h load_frag(const _Float16* p0, const _Float16* p1) {
  union { v16h v; uint4 q[2]; } u;
  u.q[0] = *(const uint4*)p0;
  u.q[1] = *(const uint4*)p1;
  return u.v;
}

// One MLP layer: D(ch x edge) = Wt(A, 16x32 tiles) x Xt(B, 32x16 tiles).
// Feature chunks in LDS: At[(chunk)*EPB + row][8 halves]; chunk q holds
// channels/K 8q..8q+7 of each row. B-frag (this code's contiguous K order, same as A): lane<16 ->
// K = kt*32 + 0..15, lane>=16 -> K = kt*32 + 16..31, N = lane&15.
template<int KT>
__device__ __forceinline__ void mlp_layer(const _Float16* At, int row,
                                          const _Float16* Wp, int lane, int half,
                                          v8f acc[NT]) {
  #pragma unroll
  for (int mt = 0; mt < NT; ++mt) acc[mt] = zero8();
  #pragma unroll
  for (int kt = 0; kt < KT; ++kt) {
    const _Float16* p0 = At + ((kt * 4 + half * 2) * EPB + row) * 8;
    v16h b = load_frag(p0, p0 + EPB * 8);
    #pragma unroll
    for (int mt = 0; mt < NT; ++mt) {
      const _Float16* q = Wp + (((kt * NT + mt) * 32 + lane) << 4);
      v16h a = load_frag(q, q + 8);
      acc[mt] = __builtin_amdgcn_wmma_f32_16x16x32_f16(false, a, false, b,
                                                       (short)0, acc[mt], false, false);
    }
  }
}

// GroupNorm(16 ch == one M tile) + ReLU. Per lane: 8 channels in regs, the
// other 8 channels of the same edge live in lane^16 -> one shfl_xor(16).
__device__ __forceinline__ void gn_relu_t(v8f acc[NT],
                                          const float* __restrict__ gamma,
                                          const float* __restrict__ beta,
                                          int half) {
  #pragma unroll
  for (int mt = 0; mt < NT; ++mt) {
    float s1 = 0.0f, s2 = 0.0f;
    #pragma unroll
    for (int r = 0; r < 8; ++r) { float v = acc[mt][r]; s1 += v; s2 += v * v; }
    s1 += __shfl_xor(s1, 16, 32);
    s2 += __shfl_xor(s2, 16, 32);
    float mean = s1 * 0.0625f;
    float var  = s2 * 0.0625f - mean * mean;
    float inv  = rsqrtf(var + 1e-5f);
    const float* gp = gamma + mt * 16 + half * 8;
    const float* bp = beta  + mt * 16 + half * 8;
    #pragma unroll
    for (int r = 0; r < 8; ++r) {
      float w = (acc[mt][r] - mean) * inv * gp[r] + bp[r];
      acc[mt][r] = w > 0.0f ? w : 0.0f;
    }
  }
}

// Write activations back as chunks: lane owns 8 contiguous channels of its edge
// -> one ds_store_b128 per tile, conflict-free (16B lane stride).
__device__ __forceinline__ void store_h(const v8f acc[NT], _Float16* At,
                                        int row, int half) {
  #pragma unroll
  for (int mt = 0; mt < NT; ++mt) {
    v8h hv;
    #pragma unroll
    for (int r = 0; r < 8; ++r) hv[r] = (_Float16)acc[mt][r];
    *(v8h*)(&At[((mt * 2 + half) * EPB + row) * 8]) = hv;
  }
}

// ---------------- weight repack into WMMA A-operand layout (as W^T) ---------
// A 16x32 f16 layout: lane<16: M=lane, halves {K0..15};
// lane>=16: M=lane-16, halves {K16..31}. M = output channel.
__global__ __launch_bounds__(256) void pack_a_kernel(const float* __restrict__ W,
                                                     _Float16* __restrict__ out,
                                                     int total) {
  int u = blockIdx.x * 256 + threadIdx.x;
  if (u >= total) return;
  int j    =  u        & 15;
  int lane = (u >> 4)  & 31;
  int mt   = (u >> 9)  & 7;
  int kt   =  u >> 12;
  int k = kt * 32 + (lane < 16 ? 0 : 16) + j;
  int n = mt * 16 + (lane & 15);
  out[u] = (_Float16)W[k * 128 + n];
}

// ---------------- encoder: Linear-GN(4)-ReLU x2, Linear ---------------------
struct EncP {
  const float *w1, *b1, *g1, *be1, *w2, *b2, *g2, *be2, *w3, *b3;
};

__device__ __forceinline__ float gn16_relu(float v, float gam, float bet) {
  float s1 = v, s2 = v * v;
  #pragma unroll
  for (int m = 1; m < 16; m <<= 1) {
    s1 += __shfl_xor(s1, m, 32);
    s2 += __shfl_xor(s2, m, 32);
  }
  float mean = s1 * 0.0625f;
  float var  = s2 * 0.0625f - mean * mean;
  float r = (v - mean) * rsqrtf(var + 1e-5f) * gam + bet;
  return r > 0.0f ? r : 0.0f;
}

__global__ __launch_bounds__(256) void encoder_kernel(const float* __restrict__ x,
                                                      const float* __restrict__ pos,
                                                      EncP P0, EncP P1,
                                                      _Float16* __restrict__ hfeat,
                                                      int N) {
  __shared__ float sh [4][64];
  __shared__ float sh2[4][64];
  int tid = threadIdx.x;
  int sub = tid >> 6;
  int j   = tid & 63;
  long sg = (long)blockIdx.x * 4 + sub;
  long node = sg >> 1;
  int which = (int)(sg & 1);
  bool valid = node < N;
  long nodeC = valid ? node : (long)(N - 1);
  const float* in = (which ? pos : x) + nodeC * 4;
  const EncP P = which ? P1 : P0;

  float v = P.b1[j];
  #pragma unroll
  for (int i = 0; i < 4; ++i) v += in[i] * P.w1[i * 64 + j];
  v = gn16_relu(v, P.g1[j], P.be1[j]);
  sh[sub][j] = v;
  __syncthreads();

  v = P.b2[j];
  for (int i = 0; i < 64; ++i) v += sh[sub][i] * P.w2[i * 64 + j];
  v = gn16_relu(v, P.g2[j], P.be2[j]);
  sh2[sub][j] = v;
  __syncthreads();

  v = P.b3[j];
  for (int i = 0; i < 64; ++i) v += sh2[sub][i] * P.w3[i * 64 + j];
  if (valid) hfeat[node * 128 + which * 64 + j] = (_Float16)v;
}

// ---------------- edge MLP + scatter-max (WMMA hot kernel) ------------------
__global__ __launch_bounds__(256) void edge_mlp_scatter(
    const _Float16* __restrict__ hfeat, const int* __restrict__ ei,
    const _Float16* __restrict__ W1p, const _Float16* __restrict__ W2p,
    const _Float16* __restrict__ W3p,
    const float* __restrict__ g1, const float* __restrict__ be1,
    const float* __restrict__ g2, const float* __restrict__ be2,
    const float* __restrict__ g3, const float* __restrict__ be3,
    unsigned int* __restrict__ agg, int E) {
  // [chunk 0..31][edge 0..127][8 halves]: chunk q = feature K 8q..8q+7. 64 KB.
  __shared__ _Float16 At[32 * EPB * 8];
  const int tid = threadIdx.x, lane = tid & 31, wave = tid >> 5;
  const int half = lane >> 4, nedge = lane & 15;
  const long eBase = (long)blockIdx.x * EPB;

  // cooperative gather: [x_i | x_j - x_i] (consecutive threads -> consecutive
  // edges -> conflict-free 16B-stride LDS stores)
  for (int it = tid; it < EPB * 16; it += 256) {
    int e  = it & (EPB - 1);
    int cc = it >> 7;
    long eg = eBase + e; if (eg >= E) eg = E - 1;
    int s = ei[eg];
    int d = ei[(long)E + eg];
    v8h xi = *(const v8h*)(hfeat + (long)d * 128 + cc * 8);
    v8h xj = *(const v8h*)(hfeat + (long)s * 128 + cc * 8);
    *(v8h*)(&At[(cc * EPB + e) * 8])        = xi;
    *(v8h*)(&At[((16 + cc) * EPB + e) * 8]) = xj - xi;
  }
  __syncthreads();

  const int row = wave * 16 + nedge;       // this lane's edge within the tile
  const long eg0 = eBase + row;
  const int dstI = (eg0 < E) ? ei[(long)E + eg0] : -1;

  v8f acc[NT];

  mlp_layer<8>(At, row, W1p, lane, half, acc);   // K = 256
  gn_relu_t(acc, g1, be1, half);
  store_h(acc, At, row, half);
  __syncthreads();

  mlp_layer<4>(At, row, W2p, lane, half, acc);   // K = 128
  gn_relu_t(acc, g2, be2, half);
  store_h(acc, At, row, half);
  __syncthreads();

  mlp_layer<4>(At, row, W3p, lane, half, acc);   // K = 128
  gn_relu_t(acc, g3, be3, half);

  // scatter-max: values >= 0 -> uint bit order == float order
  if (dstI >= 0) {
    #pragma unroll
    for (int mt = 0; mt < NT; ++mt)
      #pragma unroll
      for (int r = 0; r < 8; ++r)
        atomicMax(agg + (long)dstI * 128 + mt * 16 + half * 8 + r,
                  __float_as_uint(acc[mt][r]));
  }
}

// ---------------- enc2 (2 layers) + global max pool -------------------------
__global__ __launch_bounds__(256) void node_mlp_pool(
    const float* __restrict__ agg, const int* __restrict__ batch,
    const _Float16* __restrict__ E1p, const _Float16* __restrict__ E2p,
    const float* __restrict__ g1, const float* __restrict__ be1,
    const float* __restrict__ g2, const float* __restrict__ be2,
    unsigned int* __restrict__ gbuf, int N) {
  __shared__ _Float16 At[16 * EPB * 8];   // 32 KB
  const int tid = threadIdx.x, lane = tid & 31, wave = tid >> 5;
  const int half = lane >> 4, nedge = lane & 15;
  const long nBase = (long)blockIdx.x * EPB;

  for (int it = tid; it < EPB * 16; it += 256) {
    int e  = it & (EPB - 1);
    int cc = it >> 7;
    long ng = nBase + e; if (ng >= N) ng = N - 1;
    const float* src = agg + ng * 128 + cc * 8;
    v8h o;
    #pragma unroll
    for (int k = 0; k < 8; ++k) o[k] = (_Float16)src[k];
    *(v8h*)(&At[(cc * EPB + e) * 8]) = o;
  }
  __syncthreads();

  const int row = wave * 16 + nedge;
  const long ng0 = nBase + row;
  const int batI = (ng0 < N) ? batch[ng0] : -1;

  v8f acc[NT];

  mlp_layer<4>(At, row, E1p, lane, half, acc);
  gn_relu_t(acc, g1, be1, half);
  store_h(acc, At, row, half);
  __syncthreads();

  mlp_layer<4>(At, row, E2p, lane, half, acc);
  gn_relu_t(acc, g2, be2, half);

  if (batI >= 0) {
    #pragma unroll
    for (int mt = 0; mt < NT; ++mt)
      #pragma unroll
      for (int r = 0; r < 8; ++r)
        atomicMax(gbuf + (long)batI * 128 + mt * 16 + half * 8 + r,
                  __float_as_uint(acc[mt][r]));
  }
}

// ---------------- final MLP head --------------------------------------------
__global__ __launch_bounds__(128) void fc_kernel(const float* __restrict__ gbuf,
                                                 const float* __restrict__ w1, const float* __restrict__ b1,
                                                 const float* __restrict__ w2, const float* __restrict__ b2,
                                                 const float* __restrict__ w3, const float* __restrict__ b3,
                                                 float* __restrict__ out) {
  __shared__ float gs[128], h1[128], h2[128];
  int t = threadIdx.x, row = blockIdx.x;
  gs[t] = gbuf[row * 128 + t];
  __syncthreads();
  float s = b1[t];
  for (int i = 0; i < 128; ++i) s += gs[i] * w1[i * 128 + t];
  h1[t] = s > 0.0f ? s : 0.0f;
  __syncthreads();
  s = b2[t];
  for (int i = 0; i < 128; ++i) s += h1[i] * w2[i * 128 + t];
  h2[t] = s > 0.0f ? s : 0.0f;
  __syncthreads();
  if (t < 6) {
    s = b3[t];
    for (int i = 0; i < 128; ++i) s += h2[i] * w3[i * 6 + t];
    out[row * 6 + t] = s;
  }
}

// ---------------- launch -----------------------------------------------------
extern "C" void kernel_launch(void* const* d_in, const int* in_sizes, int n_in,
                              void* d_out, int out_size, void* d_ws, size_t ws_size,
                              hipStream_t stream) {
  const float* x     = (const float*)d_in[IN_X];
  const float* pos   = (const float*)d_in[IN_POS];
  const int*   ei    = (const int*)d_in[IN_EI];
  const int*   batch = (const int*)d_in[IN_BATCH];
  const int N = in_sizes[IN_BATCH];
  const int E = in_sizes[IN_EI] / 2;
  const int B = out_size / 6;
  (void)n_in; (void)ws_size;

  auto F = [&](int i) { return (const float*)d_in[i]; };

  size_t off = 0;
  auto carve = [&](size_t bytes) {
    void* p = (char*)d_ws + off;
    off += (bytes + 255) & ~(size_t)255;
    return p;
  };
  _Float16* hfeat = (_Float16*)carve((size_t)N * 128 * 2);
  unsigned* agg   = (unsigned*)carve((size_t)N * 128 * 4);
  unsigned* gbuf  = (unsigned*)carve((size_t)B * 128 * 4);
  _Float16* W1p   = (_Float16*)carve((size_t)256 * 128 * 2);
  _Float16* W2p   = (_Float16*)carve((size_t)128 * 128 * 2);
  _Float16* W3p   = (_Float16*)carve((size_t)128 * 128 * 2);
  _Float16* E1p   = (_Float16*)carve((size_t)128 * 128 * 2);
  _Float16* E2p   = (_Float16*)carve((size_t)128 * 128 * 2);

  hipMemsetAsync(agg,  0, (size_t)N * 128 * 4, stream);
  hipMemsetAsync(gbuf, 0, (size_t)B * 128 * 4, stream);

  pack_a_kernel<<<(256 * 128) / 256, 256, 0, stream>>>(F(LOC_W1), W1p, 256 * 128);
  pack_a_kernel<<<(128 * 128) / 256, 256, 0, stream>>>(F(LOC_W2), W2p, 128 * 128);
  pack_a_kernel<<<(128 * 128) / 256, 256, 0, stream>>>(F(LOC_W3), W3p, 128 * 128);
  pack_a_kernel<<<(128 * 128) / 256, 256, 0, stream>>>(F(E2_W1),  E1p, 128 * 128);
  pack_a_kernel<<<(128 * 128) / 256, 256, 0, stream>>>(F(E2_W2),  E2p, 128 * 128);

  EncP P0 = { F(ENC_W1),  F(ENC_B1),  F(ENC_G1),  F(ENC_BE1),
              F(ENC_W2),  F(ENC_B2),  F(ENC_G2),  F(ENC_BE2),
              F(ENC_W3),  F(ENC_B3) };
  EncP P1 = { F(ENCP_W1), F(ENCP_B1), F(ENCP_G1), F(ENCP_BE1),
              F(ENCP_W2), F(ENCP_B2), F(ENCP_G2), F(ENCP_BE2),
              F(ENCP_W3), F(ENCP_B3) };
  int encBlocks = (2 * N + 3) / 4;
  encoder_kernel<<<encBlocks, 256, 0, stream>>>(x, pos, P0, P1, hfeat, N);

  edge_mlp_scatter<<<(E + EPB - 1) / EPB, 256, 0, stream>>>(
      hfeat, ei, W1p, W2p, W3p,
      F(LOC_G1), F(LOC_BE1), F(LOC_G2), F(LOC_BE2), F(LOC_G3), F(LOC_BE3),
      agg, E);

  node_mlp_pool<<<(N + EPB - 1) / EPB, 256, 0, stream>>>(
      (const float*)agg, batch, E1p, E2p,
      F(E2_G1), F(E2_BE1), F(E2_G2), F(E2_BE2), gbuf, N);

  fc_kernel<<<B, 128, 0, stream>>>((const float*)gbuf,
                                   F(FC_W1), F(FC_B1), F(FC_W2), F(FC_B2),
                                   F(FC_W3), F(FC_B3), (float*)d_out);
}
